// MambaBB_25778393711239
// MI455X (gfx1250) — compile-verified, matrix operand order checked by reading, not run on hardware
//
#include <hip/hip_runtime.h>
#include <math.h>
#include <stdint.h>
#include <stddef.h>

// ---------------- model dimensions ----------------
#define CB    4
#define CT    512
#define CN    8
#define CBT   (CB*CT)        // 2048
#define CBTN  (CBT*CN)       // 16384
#define CBN   (CB*CN)        // 32
#define CD    256            // D_MODEL
#define CDIN  512            // D_INNER
#define CDST  128            // D_STATE
#define CNHM  8              // mamba heads
#define CHD   64             // HEADDIM
#define CCONV 768            // CONV_DIM
#define CDIP  1288           // D_IN_PROJ = 2*512 + 2*128 + 8
#define CNH   4              // attn heads
#define CPAD  64             // geo feature pad
#define CHID  128            // trunk hidden

// ---------------- WMMA types ----------------
typedef __attribute__((ext_vector_type(16))) __bf16 bf16x16;
typedef __attribute__((ext_vector_type(4)))  __bf16 bf16x4;
typedef __attribute__((ext_vector_type(8)))  float  f32x8;

__device__ inline float siluf(float v) { return v / (1.f + expf(-v)); }

// =====================================================================
// FAST WMMA GEMM: C[M,N] = A[M,K] @ B[K,N] (+bias if flags&1) (+=C if flags&2)
// Requires M%64==0, K%32==0, lda%4==0, ldb%4==0 (16B-aligned float4 loads).
// Block = 128 threads = 4 waves; block tile 64x64; each wave owns 32x32
// (2x2 of 16x16 wmma tiles, 4 v_wmma per 32-K step, operands reused 2x).
// Software-pipelined: next K-tile is prefetched into registers while the
// current tile's v_wmma ops run, so global latency overlaps compute.
// A staged as As[m][k]; B staged TRANSPOSED as Bt[n][k] so both fragments
// load as two ds_load_b128 per the documented 16-bit wave32 layout.
// =====================================================================
__global__ __launch_bounds__(128)
void k_gemm_f(const float* __restrict__ A, const float* __restrict__ Bm,
              float* __restrict__ C, const float* __restrict__ bias,
              int M, int N, int K, int lda, int ldb, int ldc, int flags)
{
  __shared__ __bf16 As[64][40];
  __shared__ __bf16 Bt[64][40];
  const int tid  = threadIdx.x;
  const int lane = tid & 31;
  const int w    = tid >> 5;
  const int wm   = w & 1, wn = w >> 1;
  const int m0   = blockIdx.y * 64;
  const int n0   = blockIdx.x * 64;

  const int ar = tid >> 1;          // 0..63  (A stage row)
  const int ac = (tid & 1) * 16;    // 0/16   (A stage col base)
  const int kr = tid >> 2;          // 0..31  (B stage k row)
  const int nb = (tid & 3) * 16;    // 0..48  (B stage n base)

  const bool nfull = (n0 + 64 <= N);   // block-uniform: whole B tile in-bounds

  f32x8  acc[2][2] = {};
  float4 ra[4], rb[4];

  auto loadA = [&](int k0) {
    const float* ap = A + (size_t)(m0 + ar) * lda + k0 + ac;
    #pragma unroll
    for (int u = 0; u < 4; ++u) ra[u] = *(const float4*)(ap + 4 * u);
  };
  auto loadB = [&](int k0) {
    const float* bp = Bm + (size_t)(k0 + kr) * ldb + n0 + nb;
    if (nfull) {
      #pragma unroll
      for (int u = 0; u < 4; ++u) rb[u] = *(const float4*)(bp + 4 * u);
    } else {
      #pragma unroll
      for (int u = 0; u < 4; ++u) {
        int c = n0 + nb + 4 * u;
        rb[u].x = (c + 0 < N) ? bp[4 * u + 0] : 0.f;
        rb[u].y = (c + 1 < N) ? bp[4 * u + 1] : 0.f;
        rb[u].z = (c + 2 < N) ? bp[4 * u + 2] : 0.f;
        rb[u].w = (c + 3 < N) ? bp[4 * u + 3] : 0.f;
      }
    }
  };
  auto stage = [&]() {
    #pragma unroll
    for (int u = 0; u < 4; ++u) {
      bf16x4 p;
      p[0] = (__bf16)ra[u].x; p[1] = (__bf16)ra[u].y;
      p[2] = (__bf16)ra[u].z; p[3] = (__bf16)ra[u].w;
      *(bf16x4*)&As[ar][ac + 4 * u] = p;
      Bt[nb + 4 * u + 0][kr] = (__bf16)rb[u].x;
      Bt[nb + 4 * u + 1][kr] = (__bf16)rb[u].y;
      Bt[nb + 4 * u + 2][kr] = (__bf16)rb[u].z;
      Bt[nb + 4 * u + 3][kr] = (__bf16)rb[u].w;
    }
  };

  loadA(0); loadB(0);

  for (int k0 = 0; k0 < K; k0 += 32) {
    stage();
    __syncthreads();
    if (k0 + 32 < K) { loadA(k0 + 32); loadB(k0 + 32); }  // overlaps wmma below

    // ---- fragments: two b128 LDS loads each ----
    const int mrow = lane & 15;
    const int kb   = (lane < 16) ? 0 : 8;
    union FragU { bf16x16 v; uint4 q[2]; };
    FragU fa[2], fb[2];
    #pragma unroll
    for (int a = 0; a < 2; ++a) {
      const __bf16* ab = &As[wm * 32 + a * 16 + mrow][0];
      fa[a].q[0] = *(const uint4*)(ab + kb);
      fa[a].q[1] = *(const uint4*)(ab + 16 + kb);
      const __bf16* bb = &Bt[wn * 32 + a * 16 + mrow][0];
      fb[a].q[0] = *(const uint4*)(bb + kb);
      fb[a].q[1] = *(const uint4*)(bb + 16 + kb);
    }
    #pragma unroll
    for (int a = 0; a < 2; ++a)
      #pragma unroll
      for (int b2 = 0; b2 < 2; ++b2)
        acc[a][b2] = __builtin_amdgcn_wmma_f32_16x16x32_bf16(
            false, fa[a].v, false, fb[b2].v, (short)0, acc[a][b2], false, false);
    __syncthreads();
  }

  // ---- store (C layout: lane n = lane&15, VGPR r -> M = r + 8*(lane>=16)) ----
  const int ncol = lane & 15;
  const int mb   = (lane < 16) ? 0 : 8;
  #pragma unroll
  for (int b2 = 0; b2 < 2; ++b2) {
    int col = n0 + wn * 32 + b2 * 16 + ncol;
    if (col >= N) continue;
    float bv = (flags & 1) ? bias[col] : 0.f;
    #pragma unroll
    for (int a = 0; a < 2; ++a) {
      int rowb = m0 + wm * 32 + a * 16 + mb;
      #pragma unroll
      for (int r = 0; r < 8; ++r) {
        size_t o = (size_t)(rowb + r) * ldc + col;
        float vv = acc[a][b2][r] + bv;
        if (flags & 2) vv += C[o];
        C[o] = vv;
      }
    }
  }
}

// =====================================================================
// SLOW/SAFE WMMA GEMM for odd shapes (K=16 embed, M=1 fold row).
// Block = 128 threads = 4 waves, 32x32 tile, full bounds checks.
// =====================================================================
__global__ __launch_bounds__(128)
void k_gemm_s(const float* __restrict__ A, const float* __restrict__ Bm,
              float* __restrict__ C, const float* __restrict__ bias,
              int M, int N, int K, int lda, int ldb, int ldc, int flags)
{
  __shared__ __bf16 As[32][36];
  __shared__ __bf16 Bs[32][36];
  const int tid  = threadIdx.x;
  const int lane = tid & 31;
  const int w    = tid >> 5;
  const int wm   = w & 1, wn = w >> 1;
  const int m0   = blockIdx.y * 32;
  const int n0   = blockIdx.x * 32;
  const int lr   = tid >> 2;
  const int lc   = (tid & 3) * 8;

  f32x8 acc = {};

  for (int k0 = 0; k0 < K; k0 += 32) {
    #pragma unroll
    for (int u = 0; u < 8; ++u) {
      int gr = m0 + lr, gc = k0 + lc + u;
      As[lr][lc + u] = (gr < M && gc < K) ? (__bf16)A[(size_t)gr * lda + gc] : (__bf16)0.f;
      int br = k0 + lr, bc = n0 + lc + u;
      Bs[lr][lc + u] = (br < K && bc < N) ? (__bf16)Bm[(size_t)br * ldb + bc] : (__bf16)0.f;
    }
    __syncthreads();
    bf16x16 fa, fb;
    const int mrow = lane & 15;
    const int kb   = (lane < 16) ? 0 : 8;
    #pragma unroll
    for (int v = 0; v < 8; ++v) {
      int kk = (v < 4) ? (kb + 2 * v) : (16 + kb + 2 * (v - 4));
      fa[2 * v]     = As[wm * 16 + mrow][kk];
      fa[2 * v + 1] = As[wm * 16 + mrow][kk + 1];
      fb[2 * v]     = Bs[kk][wn * 16 + mrow];
      fb[2 * v + 1] = Bs[kk + 1][wn * 16 + mrow];
    }
    acc = __builtin_amdgcn_wmma_f32_16x16x32_bf16(false, fa, false, fb,
                                                  (short)0, acc, false, false);
    __syncthreads();
  }

  const int ncol = lane & 15;
  const int mb   = (lane < 16) ? 0 : 8;
  const int col  = n0 + wn * 16 + ncol;
  if (col < N) {
    float bv = (flags & 1) ? bias[col] : 0.f;
    #pragma unroll
    for (int r = 0; r < 8; ++r) {
      int row = m0 + wm * 16 + mb + r;
      if (row < M) {
        float o = acc[r] + bv;
        if (flags & 2) o += C[(size_t)row * ldc + col];
        C[(size_t)row * ldc + col] = o;
      }
    }
  }
}

// ---------------- RMSNorm (+ optional fused SiLU), one wave per row ----------------
__global__ __launch_bounds__(256)
void k_rms(const float* __restrict__ X, const float* __restrict__ w,
           float* __restrict__ Y, int rows, int dim, int dosilu)
{
  int row  = blockIdx.x * 8 + (threadIdx.x >> 5);
  int lane = threadIdx.x & 31;
  if (row >= rows) return;
  const float* xr = X + (size_t)row * dim;
  float s = 0.f;
  for (int c = lane; c < dim; c += 32) { float v = xr[c]; s += v * v; }
  #pragma unroll
  for (int o = 16; o > 0; o >>= 1) s += __shfl_xor(s, o);
  float scale = rsqrtf(s / (float)dim + 1e-5f);
  float* yr = Y + (size_t)row * dim;
  for (int c = lane; c < dim; c += 32) {
    float v = xr[c] * scale * w[c];
    if (dosilu) v = siluf(v);
    yr[c] = v;
  }
}

// ---------------- elementwise / data movement kernels ----------------
__global__ void k_reset(const int* __restrict__ seq, float* __restrict__ r, int total) {
  int idx = blockIdx.x * blockDim.x + threadIdx.x; if (idx >= total) return;
  int t = idx % CT, b = idx / CT;
  r[idx] = (t > 0 && seq[b * CT + t] != seq[b * CT + t - 1]) ? 1.f : 0.f;
}

__global__ void k_prep_state(const float* __restrict__ st, const unsigned char* __restrict__ alive,
                             const float* __restrict__ dead, float* __restrict__ smask,
                             float* __restrict__ snp, int total) {
  int idx = blockIdx.x * blockDim.x + threadIdx.x; if (idx >= total) return;
  int d = idx & 15, row = idx >> 4;
  float v = alive[row] ? st[idx] : dead[d];
  smask[idx] = v;
  snp[idx] = (d == 3 || d == 4) ? 0.f : v;
}

__global__ void k_add_reset_emb(float* __restrict__ semb, const float* __restrict__ resetf,
                                const float* __restrict__ remb, int total) {
  int idx = blockIdx.x * blockDim.x + threadIdx.x; if (idx >= total) return;
  int d = idx & 255, row = idx >> 8;
  int bt = row >> 3;
  semb[idx] += resetf[bt] * remb[d];
}

__global__ void k_geo(const float* __restrict__ pos, const float* __restrict__ vel,
                      float* __restrict__ raw, int total) {
  int idx = blockIdx.x * blockDim.x + threadIdx.x; if (idx >= total) return;
  int j = idx & 7, t = idx >> 3, i = t & 7, bt = t >> 3;
  const float* pi = pos + (size_t)(bt * CN + i) * 2;
  const float* pj = pos + (size_t)(bt * CN + j) * 2;
  const float* vi = vel + (size_t)(bt * CN + i) * 2;
  const float* vj = vel + (size_t)(bt * CN + j) * 2;
  float dpx = pi[0] - pj[0], dpy = pi[1] - pj[1];
  dpx -= roundf(dpx * (1.f / 1024.f)) * 1024.f;
  dpy -= roundf(dpy * (1.f / 1024.f)) * 1024.f;
  float dvx = vi[0] - vj[0], dvy = vi[1] - vj[1];
  float dist = sqrtf(dpx * dpx + dpy * dpy + 1e-12f) + 1e-6f;
  float* o = raw + (size_t)idx * CPAD;
  o[0] = dpx; o[1] = dpy; o[2] = dvx; o[3] = dvy;
  o[4] = dist; o[5] = 1.f / (dist + 0.1f);
  o[6] = sqrtf(dvx * dvx + dvy * dvy + 1e-12f);
  o[7] = (dvx * dpx + dvy * dpy) / dist;
  o[8] = dpx / dist; o[9] = dpy / dist;
  o[10] = logf(dist + 1.f);
  const float c0 = 6.283185307179586f / 1024.f;
  float ax = dpx * c0, ay = dpy * c0;
  #pragma unroll
  for (int band = 0; band < 8; ++band) {
    float fs = (float)(1 << band);
    o[11 + band * 4 + 0] = sinf(ax * fs);
    o[11 + band * 4 + 1] = sinf(ay * fs);
    o[11 + band * 4 + 2] = cosf(ax * fs);
    o[11 + band * 4 + 3] = cosf(ay * fs);
  }
  for (int f = 43; f < 64; ++f) o[f] = 0.f;
}

__global__ void k_gather(const int* __restrict__ pa, const float* __restrict__ epow,
                         const float* __restrict__ eturn, const float* __restrict__ eshoot,
                         float* __restrict__ out, int total) {
  int idx = blockIdx.x * blockDim.x + threadIdx.x; if (idx >= total) return;
  int d = idx % CHID, row = idx / CHID;
  const int* p = pa + (size_t)row * 3;
  float v;
  if (d < 32)      v = epow[p[0] * 32 + d];
  else if (d < 96) v = eturn[p[1] * 64 + (d - 32)];
  else             v = eshoot[p[2] * 32 + (d - 96)];
  out[idx] = v;
}

// [B,T,N,D] -> [B,N,T,D]
__global__ void k_t_fw(const float* __restrict__ in, float* __restrict__ out, int total) {
  int idx = blockIdx.x * blockDim.x + threadIdx.x; if (idx >= total) return;
  int d = idx & 255, row = idx >> 8;
  int n = row & 7, bt = row >> 3, t = bt & 511, b = bt >> 9;
  out[(((size_t)(b * CN + n) * CT + t) << 8) + d] = in[idx];
}
// [B,N,T,D] -> [B,T,N,D]
__global__ void k_t_bw(const float* __restrict__ in, float* __restrict__ out, int total) {
  int idx = blockIdx.x * blockDim.x + threadIdx.x; if (idx >= total) return;
  int d = idx & 255, row = idx >> 8;
  int t = row & 511, bn = row >> 9, n = bn & 7, b = bn >> 3;
  out[(((size_t)(b * CT + t) * CN + n) << 8) + d] = in[idx];
}

__global__ void k_dtsp(const float* __restrict__ zx, const float* __restrict__ dtb,
                       float* __restrict__ dt, int total) {
  int idx = blockIdx.x * blockDim.x + threadIdx.x; if (idx >= total) return;
  int h = idx & 7; size_t row = (size_t)(idx >> 3);
  float v = zx[row * CDIP + 1280 + h] + dtb[h];
  dt[idx] = (v > 20.f) ? v : log1pf(expf(v));
}

__global__ void k_conv(const float* __restrict__ zx, const float* __restrict__ wconv,
                       const float* __restrict__ bconv, float* __restrict__ out, int total) {
  int idx = blockIdx.x * blockDim.x + threadIdx.x; if (idx >= total) return;
  int c = idx % CCONV, row = idx / CCONV;
  int t = row % CT, bn = row / CT;
  float acc = bconv[c];
  const float* wr = wconv + (size_t)c * 4;
  #pragma unroll
  for (int k2 = 0; k2 < 4; ++k2) {
    int tt = t - 3 + k2;
    if (tt >= 0) acc += wr[k2] * zx[(size_t)(bn * CT + tt) * CDIP + 512 + c];
  }
  out[idx] = siluf(acc);
}

// sequential selective scan: one block per (bn, head); h state in VGPRs
__global__ __launch_bounds__(256)
void k_scan(const float* __restrict__ xbc, const float* __restrict__ dt,
            const float* __restrict__ resetf, const float* __restrict__ A_log,
            const float* __restrict__ Dp, float* __restrict__ y)
{
  int bn = blockIdx.x >> 3;
  int h  = blockIdx.x & 7;
  int tid = threadIdx.x;
  int p = tid >> 2, sq = tid & 3, s0 = sq * 32;
  float Ac = -expf(A_log[h]);
  float Dv = Dp[h];
  int b = bn >> 3;
  __shared__ float sx[64], sB[128], sC[128];
  float hreg[32];
  #pragma unroll
  for (int j = 0; j < 32; ++j) hreg[j] = 0.f;
  for (int t = 0; t < CT; ++t) {
    size_t row = (size_t)bn * CT + t;
    const float* xr = xbc + row * CCONV;
    if (tid < 64)  sx[tid] = xr[h * CHD + tid];
    if (tid < 128) sB[tid] = xr[512 + tid];
    else           sC[tid - 128] = xr[640 + (tid - 128)];
    __syncthreads();
    if (resetf[b * CT + t] != 0.f) {
      #pragma unroll
      for (int j = 0; j < 32; ++j) hreg[j] = 0.f;
    }
    float dts  = dt[row * CNHM + h];
    float da   = expf(dts * Ac);
    float coef = dts * sx[p];
    float part = 0.f;
    #pragma unroll
    for (int j = 0; j < 32; ++j) {
      hreg[j] = da * hreg[j] + coef * sB[s0 + j];
      part += hreg[j] * sC[s0 + j];
    }
    part += __shfl_xor(part, 1);
    part += __shfl_xor(part, 2);
    if (sq == 0) y[row * CDIN + h * CHD + p] = part + Dv * sx[p];
    __syncthreads();
  }
}

__global__ void k_gated(const float* __restrict__ y, const float* __restrict__ zx,
                        float* __restrict__ out, int total) {
  int idx = blockIdx.x * blockDim.x + threadIdx.x; if (idx >= total) return;
  int c = idx & 511; size_t row = (size_t)(idx >> 9);
  float z = zx[row * CDIP + c];
  out[idx] = y[idx] * siluf(z);
}

// tiny-N attention (8 agents, 4 heads): one thread per (bt, head, query-row)
__global__ __launch_bounds__(256)
void k_attn(const float* __restrict__ qkv, const float* __restrict__ bias4,
            const unsigned char* __restrict__ alive, float* __restrict__ out, int total)
{
  int idx = blockIdx.x * blockDim.x + threadIdx.x; if (idx >= total) return;
  int n = idx & 7, t = idx >> 3, h = t & 3, bt = t >> 2;
  const float* qp = qkv + (size_t)(bt * CN + n) * 768 + h * CHD;
  float q[CHD];
  #pragma unroll
  for (int d = 0; d < CHD; ++d) q[d] = qp[d];
  float sc[CN]; float mx = -1e30f;
  for (int m = 0; m < CN; ++m) {
    const float* kp = qkv + (size_t)(bt * CN + m) * 768 + 256 + h * CHD;
    float dot = 0.f;
    #pragma unroll
    for (int d = 0; d < CHD; ++d) dot += q[d] * kp[d];
    float s = dot * 0.125f + bias4[((size_t)(bt * CN + n) * CN + m) * CNH + h];
    if (!alive[bt * CN + m]) s = -1e30f;
    sc[m] = s; mx = fmaxf(mx, s);
  }
  float sum = 0.f;
  for (int m = 0; m < CN; ++m) { sc[m] = expf(sc[m] - mx); sum += sc[m]; }
  float isum = 1.f / sum;
  float acc[CHD];
  #pragma unroll
  for (int d = 0; d < CHD; ++d) acc[d] = 0.f;
  for (int m = 0; m < CN; ++m) {
    float wm = sc[m] * isum;
    const float* vp = qkv + (size_t)(bt * CN + m) * 768 + 512 + h * CHD;
    #pragma unroll
    for (int d = 0; d < CHD; ++d) acc[d] += wm * vp[d];
  }
  float* op = out + (size_t)(bt * CN + n) * CD + h * CHD;
  #pragma unroll
  for (int d = 0; d < CHD; ++d) op[d] = acc[d];
}

__global__ void k_history(const float* __restrict__ xs, const float* __restrict__ resetf,
                          float* __restrict__ hist, int total) {
  int idx = blockIdx.x * blockDim.x + threadIdx.x; if (idx >= total) return;
  int d = idx & 255, row = idx >> 8;
  int n = row & 7, bt = row >> 3, t = bt & 511, b = bt >> 9;
  float v = 0.f;
  if (t > 0 && resetf[b * CT + t] == 0.f)
    v = xs[(((size_t)(b * CT + t - 1) * CN + n) << 8) + d];
  hist[idx] = v;
}

__global__ void k_addstate(const float* __restrict__ smask, const float* __restrict__ delta,
                           float* __restrict__ out, int total) {
  int idx = blockIdx.x * blockDim.x + threadIdx.x; if (idx >= total) return;
  out[idx] = smask[idx] + delta[idx];
}

// ---------------- host-side helpers ----------------
static void gemm(hipStream_t st, const float* A, const float* B, float* C,
                 const float* bias, int M, int N, int K, int lda, int ldb, int ldc, int flags)
{
  if ((M % 64 == 0) && (K % 32 == 0) && (lda % 4 == 0) && (ldb % 4 == 0)) {
    dim3 g((unsigned)((N + 63) / 64), (unsigned)(M / 64));
    k_gemm_f<<<g, dim3(128), 0, st>>>(A, B, C, bias, M, N, K, lda, ldb, ldc, flags);
  } else {
    dim3 g((unsigned)((N + 31) / 32), (unsigned)((M + 31) / 32));
    k_gemm_s<<<g, dim3(128), 0, st>>>(A, B, C, bias, M, N, K, lda, ldb, ldc, flags);
  }
}
static void rms(hipStream_t st, const float* x, const float* w, float* y,
                int rows, int dim, int dosilu)
{
  k_rms<<<dim3((unsigned)((rows + 7) / 8)), dim3(256), 0, st>>>(x, w, y, rows, dim, dosilu);
}
#define EL(kern, n, ...) kern<<<dim3((unsigned)(((n) + 255) / 256)), dim3(256), 0, stream>>>(__VA_ARGS__)

extern "C" void kernel_launch(void* const* d_in, const int* in_sizes, int n_in,
                              void* d_out, int out_size, void* d_ws, size_t ws_size,
                              hipStream_t stream)
{
  (void)in_sizes; (void)n_in; (void)out_size; (void)ws_size;

  // raw inputs
  const float*         state = (const float*)d_in[0];
  const int*           pact  = (const int*)d_in[1];
  const float*         pos   = (const float*)d_in[2];
  const float*         vel   = (const float*)d_in[3];
  const int*           seq   = (const int*)d_in[4];
  const unsigned char* alive = (const unsigned char*)d_in[5];
  // params (jax-tree alphabetical flatten of the params dict)
  const float* p_aa_wbias = (const float*)d_in[6];
  const float* p_aa_wproj = (const float*)d_in[7];
  const float* p_aa_wqkv  = (const float*)d_in[8];
  const float* p_aa_bproj = (const float*)d_in[9];
  const float* p_aa_bqkv  = (const float*)d_in[10];
  const float* p_af_w   = (const float*)d_in[11];
  const float* p_af_b   = (const float*)d_in[12];
  const float* p_af_rms = (const float*)d_in[13];
  const float* p_ah_w1  = (const float*)d_in[14];
  const float* p_ah_w2  = (const float*)d_in[15];
  const float* p_ah_b1  = (const float*)d_in[16];
  const float* p_ah_b2  = (const float*)d_in[17];
  const float* p_ah_rms = (const float*)d_in[18];
  const float* p_adw    = (const float*)d_in[19];
  const float* p_adb    = (const float*)d_in[20];
  const float* b_alog   = (const float*)d_in[21];
  const float* b_dp     = (const float*)d_in[22];
  const float* b_wbias  = (const float*)d_in[23];
  const float* b_wproj  = (const float*)d_in[24];
  const float* b_wqkv   = (const float*)d_in[25];
  const float* b_bproj  = (const float*)d_in[26];
  const float* b_bqkv   = (const float*)d_in[27];
  const float* b_convb  = (const float*)d_in[28];
  const float* b_convw  = (const float*)d_in[29];
  const float* b_dtbias = (const float*)d_in[30];
  const float* b_inproj = (const float*)d_in[31];
  const float* b_mnorm  = (const float*)d_in[32];
  const float* b_norm1  = (const float*)d_in[33];
  const float* b_norm2  = (const float*)d_in[34];
  const float* b_outproj= (const float*)d_in[35];
  const float* p_dead   = (const float*)d_in[36];
  const float* p_epow   = (const float*)d_in[37];
  const float* p_eshoot = (const float*)d_in[38];
  const float* p_eturn  = (const float*)d_in[39];
  const float* p_fu_w   = (const float*)d_in[40];
  const float* p_fu_b   = (const float*)d_in[41];
  const float* p_fu_rms = (const float*)d_in[42];
  const float* p_remb   = (const float*)d_in[43];
  const float* p_se_w1  = (const float*)d_in[44];
  const float* p_se_w2  = (const float*)d_in[45];
  const float* p_se_b1  = (const float*)d_in[46];
  const float* p_se_b2  = (const float*)d_in[47];
  const float* p_se_rms = (const float*)d_in[48];
  const float* p_tr_w1  = (const float*)d_in[49];
  const float* p_tr_w2  = (const float*)d_in[50];
  const float* p_tr_b1  = (const float*)d_in[51];
  const float* p_tr_b2  = (const float*)d_in[52];
  const float* p_tr_rms = (const float*)d_in[53];
  const float* p_wh_w1  = (const float*)d_in[54];
  const float* p_wh_w2  = (const float*)d_in[55];
  const float* p_wh_b1  = (const float*)d_in[56];
  const float* p_wh_b2  = (const float*)d_in[57];
  const float* p_wh_rms = (const float*)d_in[58];

  // ---- workspace arena (deterministic bump allocator) ----
  float* Wb = (float*)d_ws;
  size_t off = 0;
  auto bump = [&](size_t n) { float* p = Wb + off; off += (n + 63) & ~(size_t)63; return p; };
  float* f_reset = bump(CBT);
  float* f_snp   = bump((size_t)CBTN * 16);
  float* f_smask = bump((size_t)CBTN * 16);
  float* f_semb  = bump((size_t)CBTN * CD);
  float* f_trunk = bump((size_t)CBT * CN * CN * CHID);
  float* f_aemb  = bump((size_t)CBTN * CHID);
  float* f_x     = bump((size_t)CBTN * CD);
  float* f_xs    = bump((size_t)CBTN * CD);
  float* f_xn    = bump((size_t)CBTN * CD);
  float* f_tmp   = bump((size_t)CBTN * CD);
  float* f_zx    = bump((size_t)CBTN * CDIP);
  float* f_dt    = bump((size_t)CBTN * CNHM);
  float* f_xbc   = bump((size_t)CBTN * CCONV);
  float* f_y     = bump((size_t)CBTN * CDIN);
  float* f_y2    = bump((size_t)CBTN * CDIN);
  float* f_qkv   = bump((size_t)CBTN * 768);
  float* f_attnv = bump((size_t)CBTN * CD);
  float* f_bias4 = bump((size_t)CBT * CN * CN * CNH);
  float* f_fold  = bump(1024);
  float* f_hist  = bump((size_t)CBTN * CD);
  float* f_xa    = bump((size_t)CBTN * CD);
  float* f_delta = bump((size_t)CBTN * 16);
  float* f_raw = f_zx;   // phase-A alias (8.4M <= 21.1M)
  float* f_t1  = f_y;    // phase-A alias (16.8M == y + y2)
  float* out_pred   = (float*)d_out;
  float* out_logits = (float*)d_out + (size_t)CBTN * 16;

  const int NPAIR = CBT * CN * CN; // 131072

  // ---- phase A: embeddings, geo trunk, fusion ----
  EL(k_reset, CBT, seq, f_reset, CBT);
  EL(k_prep_state, CBTN * 16, state, alive, p_dead, f_smask, f_snp, CBTN * 16);
  gemm(stream, f_snp, p_se_w1, f_tmp, p_se_b1, CBTN, CD, 16, 16, CD, CD, 1);
  rms(stream, f_tmp, p_se_rms, f_tmp, CBTN, CD, 1);
  gemm(stream, f_tmp, p_se_w2, f_semb, p_se_b2, CBTN, CD, CD, CD, CD, CD, 1);
  EL(k_add_reset_emb, CBTN * CD, f_semb, f_reset, p_remb, CBTN * CD);
  EL(k_geo, NPAIR, pos, vel, f_raw, NPAIR);
  gemm(stream, f_raw, p_tr_w1, f_t1, p_tr_b1, NPAIR, CHID, CPAD, CPAD, CHID, CHID, 1);
  rms(stream, f_t1, p_tr_rms, f_t1, NPAIR, CHID, 1);
  gemm(stream, f_t1, p_tr_w2, f_trunk, p_tr_b2, NPAIR, CHID, CHID, CHID, CHID, CHID, 1);
  EL(k_gather, CBTN * CHID, pact, p_epow, p_eturn, p_eshoot, f_aemb, CBTN * CHID);
  gemm(stream, f_semb, p_fu_w, f_xs, p_fu_b, CBTN, CD, CD, CD, CD, CD, 1);
  gemm(stream, f_aemb, p_fu_w + (size_t)CD * CD, f_xs, nullptr, CBTN, CD, CHID, CHID, CD, CD, 2);
  rms(stream, f_xs, p_fu_rms, f_xs, CBTN, CD, 1);
  EL(k_t_fw, CBTN * CD, f_xs, f_x, CBTN * CD);   // -> [B,N,T,D]

  // ---- phase B: 3 (mamba + rel-attn) blocks ----
  for (int l = 0; l < 3; ++l) {
    const float* inproj  = b_inproj  + (size_t)l * CD * CDIP;
    const float* outproj = b_outproj + (size_t)l * CDIN * CD;
    const float* wqkv    = b_wqkv    + (size_t)l * CD * 768;
    const float* bqkv    = b_bqkv    + (size_t)l * 768;
    const float* wproj   = b_wproj   + (size_t)l * CD * CD;
    const float* bproj   = b_bproj   + (size_t)l * CD;
    const float* wbias   = b_wbias   + (size_t)l * CD * CNH;
    const float* convw   = b_convw   + (size_t)l * CCONV * 4;
    const float* convb   = b_convb   + (size_t)l * CCONV;
    const float* dtbias  = b_dtbias  + (size_t)l * CNHM;
    const float* alog    = b_alog    + (size_t)l * CNHM;
    const float* dp      = b_dp      + (size_t)l * CNHM;
    const float* norm1   = b_norm1   + (size_t)l * CD;
    const float* norm2   = b_norm2   + (size_t)l * CD;
    const float* mnorm   = b_mnorm   + (size_t)l * CDIN;

    // Mamba2 (rows = (bn, t))
    rms(stream, f_x, norm1, f_xn, CBTN, CD, 0);
    gemm(stream, f_xn, inproj, f_zx, nullptr, CBTN, CDIP, CD, CD, CDIP, CDIP, 0);
    EL(k_dtsp, CBTN * CNHM, f_zx, dtbias, f_dt, CBTN * CNHM);
    EL(k_conv, CBTN * CCONV, f_zx, convw, convb, f_xbc, CBTN * CCONV);
    k_scan<<<dim3(CBN * CNHM), dim3(256), 0, stream>>>(f_xbc, f_dt, f_reset, alog, dp, f_y);
    EL(k_gated, CBTN * CDIN, f_y, f_zx, f_y2, CBTN * CDIN);
    rms(stream, f_y2, mnorm, f_y2, CBTN, CDIN, 0);
    gemm(stream, f_y2, outproj, f_x, nullptr, CBTN, CD, CDIN, CDIN, CD, CD, 2); // residual add

    // relational attention (rows = (b, t, n))
    EL(k_t_bw, CBTN * CD, f_x, f_xs, CBTN * CD);
    rms(stream, f_xs, norm2, f_xn, CBTN, CD, 0);
    gemm(stream, f_xn, wqkv, f_qkv, bqkv, CBTN, 768, CD, CD, 768, 768, 1);
    // fold (ad_W[l+1] @ Wbias, ad_b[l+1] @ Wbias) -> [128,4] + [4]
    gemm(stream, p_adw + (size_t)(l + 1) * CHID * CD, wbias, f_fold, nullptr,
         CHID, CNH, CD, CD, CNH, CNH, 0);
    gemm(stream, p_adb + (size_t)(l + 1) * CD, wbias, f_fold + 512, nullptr,
         1, CNH, CD, CD, CNH, CNH, 0);
    gemm(stream, f_trunk, f_fold, f_bias4, f_fold + 512, NPAIR, CNH, CHID, CHID, CNH, CNH, 1);
    EL(k_attn, CBT * CNH * CN, f_qkv, f_bias4, alive, f_attnv, CBT * CNH * CN);
    gemm(stream, f_attnv, wproj, f_xs, bproj, CBTN, CD, CD, CD, CD, CD, 3); // bias + residual
    EL(k_t_fw, CBTN * CD, f_xs, f_x, CBTN * CD);
  }
  // f_xs now holds x_final in [B,T,N,D]

  // ---- world head -> state_pred ----
  gemm(stream, f_xs, p_wh_w1, f_tmp, p_wh_b1, CBTN, CD, CD, CD, CD, CD, 1);
  rms(stream, f_tmp, p_wh_rms, f_tmp, CBTN, CD, 1);
  gemm(stream, f_tmp, p_wh_w2, f_delta, p_wh_b2, CBTN, 16, CD, CD, 16, 16, 1);
  EL(k_addstate, CBTN * 16, f_smask, f_delta, out_pred, CBTN * 16);

  // ---- actor branch -> logits ----
  EL(k_history, CBTN * CD, f_xs, f_reset, f_hist, CBTN * CD);
  gemm(stream, f_semb, p_af_w, f_xa, p_af_b, CBTN, CD, CD, CD, CD, CD, 1);
  gemm(stream, f_hist, p_af_w + (size_t)CD * CD, f_xa, nullptr, CBTN, CD, CD, CD, CD, CD, 2);
  rms(stream, f_xa, p_af_rms, f_xa, CBTN, CD, 1);
  gemm(stream, f_xa, p_aa_wqkv, f_qkv, p_aa_bqkv, CBTN, 768, CD, CD, 768, 768, 1);
  gemm(stream, p_adw, p_aa_wbias, f_fold, nullptr, CHID, CNH, CD, CD, CNH, CNH, 0);
  gemm(stream, p_adb, p_aa_wbias, f_fold + 512, nullptr, 1, CNH, CD, CD, CNH, CNH, 0);
  gemm(stream, f_trunk, f_fold, f_bias4, f_fold + 512, NPAIR, CNH, CHID, CHID, CNH, CNH, 1);
  EL(k_attn, CBT * CNH * CN, f_qkv, f_bias4, alive, f_attnv, CBT * CNH * CN);
  gemm(stream, f_attnv, p_aa_wproj, f_xn, p_aa_bproj, CBTN, CD, CD, CD, CD, CD, 1);
  gemm(stream, f_xn, p_ah_w1, f_tmp, p_ah_b1, CBTN, CD, CD, CD, CD, CD, 1);
  rms(stream, f_tmp, p_ah_rms, f_tmp, CBTN, CD, 1);
  gemm(stream, f_tmp, p_ah_w2, out_logits, p_ah_b2, CBTN, 12, CD, CD, 12, 12, 1);
}
